// AsynchronizedAttention_40982577938840
// MI455X (gfx1250) — matrix operand order repaired and checked by reading, not compiled, not run on hardware
//
#include <hip/hip_runtime.h>
#include <hip/hip_bf16.h>

// Problem constants (from reference)
#define B_    4
#define S_    4096
#define DIM_  1024
#define H_    16
#define HD_   64
#define NTOK  (B_ * S_)     // 16384 tokens
#define N3    (3 * DIM_)    // 3072 qkv columns
#define UMV_  20.0f

#define MTOK  32            // tokens per workgroup (2 row-tiles of 16)
#define NTG   4             // N-tiles per register group

typedef _Float16 v16h __attribute__((ext_vector_type(16)));
typedef _Float16 v8h  __attribute__((ext_vector_type(8)));
typedef _Float16 v4h  __attribute__((ext_vector_type(4)));
typedef float    v8f  __attribute__((ext_vector_type(8)));
typedef float    v4f  __attribute__((ext_vector_type(4)));

// ---------------------------------------------------------------------------
// Kernel 0: W [1024][3072] f32 -> Wt [3072][1024] f16 (transposed: WMMA B
// fragments become contiguous 32B loads per lane)
// ---------------------------------------------------------------------------
__global__ void wt_transpose_kernel(const float* __restrict__ W,
                                    _Float16* __restrict__ Wt) {
    __shared__ float tile[32][33];
    const int bx = blockIdx.x;   // n tile
    const int by = blockIdx.y;   // k tile
    const int tx = threadIdx.x;  // 0..31
    const int ty = threadIdx.y;  // 0..7
    const int nIn = bx * 32 + tx;
#pragma unroll
    for (int r = 0; r < 4; ++r) {
        const int k = by * 32 + ty + r * 8;
        tile[ty + r * 8][tx] = W[(size_t)k * N3 + nIn];
    }
    __syncthreads();
    const int kOut = by * 32 + tx;
#pragma unroll
    for (int r = 0; r < 4; ++r) {
        const int n = bx * 32 + ty + r * 8;
        Wt[(size_t)n * DIM_ + kOut] = (_Float16)tile[tx][ty + r * 8];
    }
}

// ---------------------------------------------------------------------------
// Kernel 1: x f32 -> f16
// ---------------------------------------------------------------------------
__global__ void xh_convert_kernel(const float* __restrict__ x,
                                  _Float16* __restrict__ xh) {
    const size_t i = ((size_t)blockIdx.x * blockDim.x + threadIdx.x) * 4;
    v4f v = *(const v4f*)(x + i);
    *(v4h*)(xh + i) = __builtin_convertvector(v, v4h);
}

// ---------------------------------------------------------------------------
// Kernel 2 (fused): 512 workgroups x 8 wave32, 32 tokens each.
//   A tile (64KB) async-copied to LDS (global_load_async_to_lds_b128).
//   Phase 1: qkv = xh @ Wt^T + bias; B double-buffered in registers so each
//            load clause is covered by 8 WMMAs before its wait.
//   Phase 2: per-token cross-head attention (QK^T, masked exp, P V).
// ---------------------------------------------------------------------------
__global__ __launch_bounds__(256)
void fused_qkv_headattn_kernel(const _Float16* __restrict__ xh,
                               const _Float16* __restrict__ Wt,
                               const float* __restrict__ bias,
                               float* __restrict__ out) {
    // LDS: 64KB A + 128KB QK + 64KB V^T + 4KB P = 260KB (WGP has 320KB)
    __shared__ __attribute__((aligned(128))) _Float16 aTile[MTOK * DIM_];
    __shared__ __attribute__((aligned(64)))  _Float16 qk[MTOK * 2048]; // [tok][h*128+(q|k)]
    __shared__ __attribute__((aligned(64)))  _Float16 vt[MTOK * 1024]; // [tok][d*16+h]
    __shared__ __attribute__((aligned(64)))  _Float16 pscr[8 * 256];   // per-wave P bounce

    const int wave  = threadIdx.x >> 5;
    const int lane  = threadIdx.x & 31;
    const int lhalf = lane >> 4;   // 0: lanes 0-15, 1: lanes 16-31
    const int lmod  = lane & 15;
    const int t0    = blockIdx.x * MTOK;

    // -------- Async copy of the 64KB A tile (32 tokens x 1024 f16) to LDS --
    {
        const unsigned ldsA0 = (unsigned)(size_t)&aTile[0];
        const unsigned long long gsrc =
            (unsigned long long)(const char*)(xh + (size_t)t0 * DIM_);
#pragma unroll
        for (int i = 0; i < 16; ++i) {   // 256 thr * 16 iters * 16B = 64KB
            const unsigned off = ((unsigned)threadIdx.x + i * 256u) * 16u;
            const unsigned ldsaddr = ldsA0 + off;
            const unsigned long long gaddr = gsrc + off;
            asm volatile("global_load_async_to_lds_b128 %0, %1, off"
                         :: "v"(ldsaddr), "v"(gaddr) : "memory");
        }
        asm volatile("s_wait_asynccnt 0" ::: "memory");
        __syncthreads();
    }

    // ---------------- Phase 1: QKV GEMM ----------------
    const _Float16* aRow0 = aTile + (size_t)lmod * DIM_ + (lhalf << 4);
    const _Float16* aRow1 = aRow0 + (size_t)16 * DIM_;     // second row-tile

    for (int g = 0; g < 24 / NTG; ++g) {
        const int nt0 = wave * 24 + g * NTG;
        v8f acc0[NTG], acc1[NTG];
        const _Float16* wcol[NTG];
#pragma unroll
        for (int j = 0; j < NTG; ++j) {
            const int col = (nt0 + j) * 16 + lmod;
            wcol[j] = Wt + (size_t)col * DIM_ + (lhalf << 4);
            const float bv = bias[col];
#pragma unroll
            for (int i = 0; i < 8; ++i) { acc0[j][i] = bv; acc1[j][i] = bv; }
        }
        // Software-pipelined B stream: ping-pong register buffers.
        v16h b0[NTG], b1[NTG];
#pragma unroll
        for (int j = 0; j < NTG; ++j) b0[j] = *(const v16h*)(wcol[j]);
        for (int kk = 0; kk < DIM_; kk += 64) {
#pragma unroll
            for (int j = 0; j < NTG; ++j)
                b1[j] = *(const v16h*)(wcol[j] + kk + 32);
            {   // WMMAs on b0 (K-chunk kk) — covers b1's load latency
                const v8h l0 = *(const v8h*)(aRow0 + kk);
                const v8h h0 = *(const v8h*)(aRow0 + kk + 8);
                const v8h l1 = *(const v8h*)(aRow1 + kk);
                const v8h h1 = *(const v8h*)(aRow1 + kk + 8);
                const v16h a0 = __builtin_shufflevector(l0, h0,
                    0, 1, 2, 3, 4, 5, 6, 7, 8, 9, 10, 11, 12, 13, 14, 15);
                const v16h a1 = __builtin_shufflevector(l1, h1,
                    0, 1, 2, 3, 4, 5, 6, 7, 8, 9, 10, 11, 12, 13, 14, 15);
#pragma unroll
                for (int j = 0; j < NTG; ++j) {
                    acc0[j] = __builtin_amdgcn_wmma_f32_16x16x32_f16(
                        false, a0, false, b0[j], (short)0, acc0[j], false, false);
                    acc1[j] = __builtin_amdgcn_wmma_f32_16x16x32_f16(
                        false, a1, false, b0[j], (short)0, acc1[j], false, false);
                }
            }
            const int kn = (kk + 64) & (DIM_ - 1);  // wraps to 0 on last iter
#pragma unroll
            for (int j = 0; j < NTG; ++j)
                b0[j] = *(const v16h*)(wcol[j] + kn);
            {   // WMMAs on b1 (K-chunk kk+32) — covers b0's load latency
                const v8h l0 = *(const v8h*)(aRow0 + kk + 32);
                const v8h h0 = *(const v8h*)(aRow0 + kk + 40);
                const v8h l1 = *(const v8h*)(aRow1 + kk + 32);
                const v8h h1 = *(const v8h*)(aRow1 + kk + 40);
                const v16h a0 = __builtin_shufflevector(l0, h0,
                    0, 1, 2, 3, 4, 5, 6, 7, 8, 9, 10, 11, 12, 13, 14, 15);
                const v16h a1 = __builtin_shufflevector(l1, h1,
                    0, 1, 2, 3, 4, 5, 6, 7, 8, 9, 10, 11, 12, 13, 14, 15);
#pragma unroll
                for (int j = 0; j < NTG; ++j) {
                    acc0[j] = __builtin_amdgcn_wmma_f32_16x16x32_f16(
                        false, a0, false, b1[j], (short)0, acc0[j], false, false);
                    acc1[j] = __builtin_amdgcn_wmma_f32_16x16x32_f16(
                        false, a1, false, b1[j], (short)0, acc1[j], false, false);
                }
            }
        }
        // Scatter D tiles to LDS staging (f16).
#pragma unroll
        for (int j = 0; j < NTG; ++j) {
            const int col = (nt0 + j) * 16 + lmod;
            const int h   = col / 192;        // uniform within a 16-col tile
            const int jm  = col - h * 192;    // 0..191
#pragma unroll
            for (int i = 0; i < 8; ++i) {
                const int r0 = i + (lhalf << 3);        // row-tile 0
                const int r1 = r0 + 16;                 // row-tile 1
                const _Float16 v0 = (_Float16)acc0[j][i];
                const _Float16 v1 = (_Float16)acc1[j][i];
                if (jm < 128) {
                    qk[r0 * 2048 + h * 128 + jm] = v0;
                    qk[r1 * 2048 + h * 128 + jm] = v1;
                } else {
                    vt[r0 * 1024 + (jm - 128) * 16 + h] = v0;
                    vt[r1 * 1024 + (jm - 128) * 16 + h] = v1;
                }
            }
        }
    }
    __syncthreads();

    // ---------------- Phase 2: cross-head attention, 4 tokens per wave -----
#pragma unroll
    for (int tt = 0; tt < MTOK / 8; ++tt) {
        const int tl = wave * (MTOK / 8) + tt;
        const _Float16* qkt = qk + tl * 2048;

        // S = Q @ K^T  (16h x 16g, contraction over head_dim=64)
        v8f s = {};
#pragma unroll
        for (int dk = 0; dk < 64; dk += 32) {
            const _Float16* qp = qkt + lmod * 128 + dk + (lhalf << 4);
            const v8h qlo = *(const v8h*)qp;
            const v8h qhi = *(const v8h*)(qp + 8);
            const v16h aq = __builtin_shufflevector(qlo, qhi,
                0, 1, 2, 3, 4, 5, 6, 7, 8, 9, 10, 11, 12, 13, 14, 15);
            const _Float16* kp = qkt + lmod * 128 + 64 + dk + (lhalf << 4);
            const v16h bk = *(const v16h*)kp;
            s = __builtin_amdgcn_wmma_f32_16x16x32_f16(
                false, aq, false, bk, (short)0, s, false, false);
        }

        // masked exp + per-row normalize (rows live in 16-lane halves;
        // xor masks 1,2,4,8 reduce over the head axis g).
        float p[8];
#pragma unroll
        for (int i = 0; i < 8; ++i) {
            const float sv = s[i] - UMV_;
            const float ev = (sv >= -UMV_ && sv <= UMV_) ? __expf(sv) : 0.0f;
            float den = ev;
            den += __shfl_xor(den, 1, 32);
            den += __shfl_xor(den, 2, 32);
            den += __shfl_xor(den, 4, 32);
            den += __shfl_xor(den, 8, 32);
            p[i] = ev / den;
        }

        // Re-lay P from C/D layout into A-fragment layout through LDS.
        _Float16* ps = pscr + wave * 256;
#pragma unroll
        for (int i = 0; i < 8; ++i) {
            const int r = i + (lhalf << 3);
            ps[r * 16 + lmod] = (_Float16)p[i];
        }
        asm volatile("s_wait_dscnt 0" ::: "memory");  // wave-local LDS RAW
        const v8h plo = *(const v8h*)(ps + lmod * 16);
        const v8h z8 = *(const v8h*)(ps + lmod * 16 + 8);
        const v16h aP = __builtin_shufflevector(plo, z8,
            0, 1, 2, 3, 4, 5, 6, 7, 8, 9, 10, 11, 12, 13, 14, 15);

        // out[h][d] = P[h][g] * V[g][d]; K zero-padded 16->32. Lanes 16-31
        // carry pad rows -> multiply by 0 (branchless, EXEC all ones).
        const _Float16 bscale = (_Float16)(lhalf ? 0.0f : 1.0f);
        const size_t obase = (size_t)(t0 + tl) * 1024;
#pragma unroll
        for (int c = 0; c < 4; ++c) {
            v16h bv = *(const v16h*)(vt + tl * 1024 + (c * 16 + lmod) * 16);
            bv = bv * bscale;
            v8f oacc = {};
            oacc = __builtin_amdgcn_wmma_f32_16x16x32_f16(
                false, aP, false, bv, (short)0, oacc, false, false);
#pragma unroll
            for (int i = 0; i < 8; ++i) {
                const int hrow = i + (lhalf << 3);
                out[obase + (size_t)hrow * 64 + c * 16 + lmod] = oacc[i];
            }
        }
    }
}

// ---------------------------------------------------------------------------
extern "C" void kernel_launch(void* const* d_in, const int* in_sizes, int n_in,
                              void* d_out, int out_size, void* d_ws, size_t ws_size,
                              hipStream_t stream) {
    const float* x    = (const float*)d_in[0];   // [16384][1024] f32
    const float* W    = (const float*)d_in[1];   // [1024][3072] f32
    const float* bias = (const float*)d_in[2];   // [3072] f32
    float* out        = (float*)d_out;           // [16384][16][64] f32

    _Float16* Wt = (_Float16*)d_ws;                                   // 6 MB
    _Float16* xh = (_Float16*)((char*)d_ws + (size_t)N3 * DIM_ * 2);  // 32 MB

    wt_transpose_kernel<<<dim3(N3 / 32, DIM_ / 32), dim3(32, 8), 0, stream>>>(W, Wt);
    xh_convert_kernel<<<(NTOK * DIM_ / 4) / 256, 256, 0, stream>>>(x, xh);
    fused_qkv_headattn_kernel<<<NTOK / MTOK, 256, 0, stream>>>(xh, Wt, bias, out);
}
